// SwinTransformerBlock_21852793602093
// MI455X (gfx1250) — matrix operand order repaired and checked by reading, not compiled, not run on hardware
//
#include <hip/hip_runtime.h>

// ---------------------------------------------------------------------------
// Swin block: B=8, H=W=128, C=192, NH=6, HD=32, WS=8, SS=4, N=64 tokens/window
// ---------------------------------------------------------------------------

typedef __attribute__((ext_vector_type(16))) _Float16 v16h;
typedef __attribute__((ext_vector_type(8)))  _Float16 v8h;
typedef __attribute__((ext_vector_type(8)))  float    v8f;

#define C_    192
#define XSTR  200   // LDS row stride (halves) for 192-wide tiles
#define ASTR  72    // LDS row stride (halves) for 64-wide (token-axis) tiles
#define HSTR  784   // LDS row stride (halves) for 768-wide MLP tiles

__device__ __forceinline__ v8f wmma_f16(v16h a, v16h b, v8f c) {
  return __builtin_amdgcn_wmma_f32_16x16x32_f16(false, a, false, b, (short)0, c,
                                                false, false);
}

// A fragment (16x32, MxK): lane&15 = row, halves e<8 -> K = kh+e, e>=8 -> kh+16+e-8
__device__ __forceinline__ v16h frag_a(const _Float16* base, int stride, int m0, int k0) {
  const int lane = threadIdx.x & 31;
  const _Float16* p = base + (m0 + (lane & 15)) * stride + k0 + ((lane >> 4) << 3);
  v8h lo = *(const v8h*)(p);
  v8h hi = *(const v8h*)(p + 16);
  v16h a;
#pragma unroll
  for (int i = 0; i < 8; ++i) { a[i] = lo[i]; a[i + 8] = hi[i]; }
  return a;
}

// B fragment (32x16, KxN) from K-contiguous storage: element (k,n) at base[n*stride+k]
__device__ __forceinline__ v16h frag_b(const _Float16* base, int stride, int n0, int k0) {
  const int lane = threadIdx.x & 31;
  const _Float16* p = base + (n0 + (lane & 15)) * stride + k0 + ((lane >> 4) << 3);
  v8h lo = *(const v8h*)(p);
  v8h hi = *(const v8h*)(p + 16);
  v16h b;
#pragma unroll
  for (int i = 0; i < 8; ++i) { b[i] = lo[i]; b[i + 8] = hi[i]; }
  return b;
}

// ---------------------------------------------------------------------------
__global__ void k_cvt(const float* __restrict__ s, _Float16* __restrict__ d, int n) {
  int i = blockIdx.x * blockDim.x + threadIdx.x;
  if (i < n) d[i] = (_Float16)s[i];
}

// bias_comb[wi][h][m][n] = attn_mask[wi][m][n] + rpb_table[rpi[m][n]][h]
__global__ void k_bias(const float* __restrict__ mask, const int* __restrict__ rpi,
                       const float* __restrict__ rpb, float* __restrict__ out) {
  int idx = blockIdx.x * blockDim.x + threadIdx.x;   // over 256*4096
  if (idx >= 256 * 4096) return;
  int wi = idx >> 12, mn = idx & 4095;
  float mv = mask[idx];
  int r = rpi[mn];
#pragma unroll
  for (int h = 0; h < 6; ++h)
    out[(((size_t)(wi * 6 + h)) << 12) + mn] = mv + rpb[r * 6 + h];
}

// ---------------------------------------------------------------------------
// Attention kernel: one block per window (2048 windows), 6 waves = 6 heads.
// LDS (halves): qb[0,12800) kb[12800,25600) vT[25600,39424) xin/scr[39424,67072)
__global__ __launch_bounds__(192)
void k_attn(const float* __restrict__ x,
            const float* __restrict__ n1g, const float* __restrict__ n1b,
            const _Float16* __restrict__ qkvw, const float* __restrict__ qkvb,
            const _Float16* __restrict__ gatew, const float* __restrict__ gateb,
            const _Float16* __restrict__ projw, const float* __restrict__ projb,
            const float* __restrict__ biasc,
            float* __restrict__ xres) {
  extern __shared__ _Float16 lds[];
  _Float16* qb  = lds;            // 64*200 (reused as gated-out buffer)
  _Float16* kb  = lds + 12800;    // 64*200
  _Float16* vt  = lds + 25600;    // 192*72  : v transposed [dim][token]
  _Float16* xin = lds + 39424;    // 64*200  (dead after qkv/gate GEMMs)
  _Float16* scr = lds + 39424;    // 6*64*72 attn-prob scratch (aliases xin)

  const int wv = threadIdx.x >> 5;      // wave id == head id (0..5)
  const int lane = threadIdx.x & 31;
  const int widx = blockIdx.x;
  const int b   = widx >> 8;
  const int wim = widx & 255;
  const int wh  = wim >> 4, ww = wim & 15;

  // ---- LN1 + cyclic-shift gather into xin (f16) ----
  for (int t = wv; t < 64; t += 6) {
    int i = t >> 3, j = t & 7;
    int h = (wh * 8 + i + 4) & 127;
    int w = (ww * 8 + j + 4) & 127;
    const float* row = x + ((size_t)(b * 16384 + h * 128 + w)) * C_;
    float v0[6];
    float s = 0.f;
#pragma unroll
    for (int kk = 0; kk < 6; ++kk) { v0[kk] = row[kk * 32 + lane]; s += v0[kk]; }
#pragma unroll
    for (int off = 16; off; off >>= 1) s += __shfl_xor(s, off, 32);
    float mean = s * (1.f / 192.f);
    float q2 = 0.f;
#pragma unroll
    for (int kk = 0; kk < 6; ++kk) { float d = v0[kk] - mean; q2 += d * d; }
#pragma unroll
    for (int off = 16; off; off >>= 1) q2 += __shfl_xor(q2, off, 32);
    float rstd = rsqrtf(q2 * (1.f / 192.f) + 1e-5f);
#pragma unroll
    for (int kk = 0; kk < 6; ++kk) {
      int c = kk * 32 + lane;
      xin[t * XSTR + c] = (_Float16)((v0[kk] - mean) * rstd * n1g[c] + n1b[c]);
    }
  }
  __syncthreads();

  // ---- qkv = xin @ qkv_w^T  ([64,576], cols: q|k|v, head-major within each) ----
  const float qscale = 0.17677669529663687f;  // HD^-0.5
  for (int tid = wv * 24; tid < wv * 24 + 24; ++tid) {
    int nt = tid >> 2, mt = tid & 3;
    v8f acc = {};
#pragma unroll
    for (int ks = 0; ks < 6; ++ks) {
      v16h a = frag_a(xin, XSTR, mt * 16, ks * 32);
      v16h bf = frag_b(qkvw, C_, nt * 16, ks * 32);
      acc = wmma_f16(a, bf, acc);
    }
    int ncol = nt * 16 + (lane & 15);
    float bias = qkvb[ncol];
    int which = ncol / 192;                    // uniform per tile (192 % 16 == 0)
    int ccol = ncol - which * 192;
    int m0 = mt * 16 + ((lane >> 4) << 3);
    if (which == 2) {
      // v: store transposed & packed -> one 16B DS store (8 tokens along a dim row)
      v8h pk;
#pragma unroll
      for (int r = 0; r < 8; ++r) pk[r] = (_Float16)(acc[r] + bias);
      *(v8h*)(vt + ccol * ASTR + m0) = pk;
    } else {
      _Float16* dst = (which == 0) ? qb : kb;
      float scale = (which == 0) ? qscale : 1.f;
#pragma unroll
      for (int r = 0; r < 8; ++r)
        dst[(m0 + r) * XSTR + ccol] = (_Float16)((acc[r] + bias) * scale);
    }
  }

  // ---- gates = xin @ gate_w^T, wave keeps its head's 32 columns in regs ----
  v8f gacc[4][2];
#pragma unroll
  for (int mt = 0; mt < 4; ++mt)
#pragma unroll
    for (int nt = 0; nt < 2; ++nt) {
      int n0 = wv * 32 + nt * 16;
      v8f acc = {};
#pragma unroll
      for (int ks = 0; ks < 6; ++ks) {
        v16h a = frag_a(xin, XSTR, mt * 16, ks * 32);
        v16h bf = frag_b(gatew, C_, n0, ks * 32);
        acc = wmma_f16(a, bf, acc);
      }
      float gbv = gateb[n0 + (lane & 15)];
#pragma unroll
      for (int r = 0; r < 8; ++r) acc[r] += gbv;
      gacc[mt][nt] = acc;
    }
  __syncthreads();  // xin dead; scr region becomes live

  // ---- scores = q @ k^T + bias(+mask), one head per wave ----
  const int head = wv;
  const float* bc = biasc + ((size_t)(wim * 6 + head)) * 4096;
  v8f att[4][4];
  {
    v16h bfr[4];
#pragma unroll
    for (int nt = 0; nt < 4; ++nt) bfr[nt] = frag_b(kb, XSTR, nt * 16, head * 32);
#pragma unroll
    for (int mt = 0; mt < 4; ++mt) {
      v16h a = frag_a(qb, XSTR, mt * 16, head * 32);
#pragma unroll
      for (int nt = 0; nt < 4; ++nt) {
        v8f acc = {};
        acc = wmma_f16(a, bfr[nt], acc);
        int n = nt * 16 + (lane & 15);
        int m0 = mt * 16 + ((lane >> 4) << 3);
#pragma unroll
        for (int r = 0; r < 8; ++r) acc[r] += bc[(m0 + r) * 64 + n];
        att[mt][nt] = acc;
      }
    }
  }

  // ---- softmax over n (rows live in 16-lane halves; xor{1,2,4,8} stays inside) ----
  _Float16* myscr = scr + wv * (64 * ASTR);
#pragma unroll
  for (int mt = 0; mt < 4; ++mt) {
#pragma unroll
    for (int r = 0; r < 8; ++r) {
      float mx = fmaxf(fmaxf(att[mt][0][r], att[mt][1][r]),
                       fmaxf(att[mt][2][r], att[mt][3][r]));
#pragma unroll
      for (int off = 8; off; off >>= 1) mx = fmaxf(mx, __shfl_xor(mx, off, 32));
      float sum = 0.f;
      float p[4];
#pragma unroll
      for (int nt = 0; nt < 4; ++nt) { p[nt] = __expf(att[mt][nt][r] - mx); sum += p[nt]; }
#pragma unroll
      for (int off = 8; off; off >>= 1) sum += __shfl_xor(sum, off, 32);
      float inv = 1.f / sum;
      int m = mt * 16 + ((lane >> 4) << 3) + r;
#pragma unroll
      for (int nt = 0; nt < 4; ++nt)
        myscr[m * ASTR + nt * 16 + (lane & 15)] = (_Float16)(p[nt] * inv);
    }
  }
  __syncthreads();  // all score reads of qb done -> qb reusable as out buffer

  // ---- out = attn @ v, gated by sigmoid(gates); store f16 into qb ----
#pragma unroll
  for (int nt = 0; nt < 2; ++nt) {
    int nb = head * 32 + nt * 16;            // dim block within vT
    v16h b0 = frag_b(vt, ASTR, nb, 0);
    v16h b1 = frag_b(vt, ASTR, nb, 32);
#pragma unroll
    for (int mt = 0; mt < 4; ++mt) {
      v8f acc = {};
      acc = wmma_f16(frag_a(myscr, ASTR, mt * 16, 0), b0, acc);
      acc = wmma_f16(frag_a(myscr, ASTR, mt * 16, 32), b1, acc);
      int m0 = mt * 16 + ((lane >> 4) << 3);
#pragma unroll
      for (int r = 0; r < 8; ++r) {
        float g = gacc[mt][nt][r];
        float val = acc[r] / (1.f + __expf(-g));
        qb[(m0 + r) * XSTR + nb + (lane & 15)] = (_Float16)val;
      }
    }
  }
  __syncthreads();

  // ---- proj + reverse shift + residual scatter into xres(d_out) ----
#pragma unroll
  for (int mt = 0; mt < 4; ++mt)
#pragma unroll
    for (int ntl = 0; ntl < 2; ++ntl) {
      int n0 = wv * 32 + ntl * 16;
      v8f acc = {};
#pragma unroll
      for (int ks = 0; ks < 6; ++ks) {
        v16h a = frag_a(qb, XSTR, mt * 16, ks * 32);
        v16h bf = frag_b(projw, C_, n0, ks * 32);
        acc = wmma_f16(a, bf, acc);
      }
      int c = n0 + (lane & 15);
      float pb = projb[c];
      int m0 = mt * 16 + ((lane >> 4) << 3);
#pragma unroll
      for (int r = 0; r < 8; ++r) {
        int m = m0 + r;
        int i = m >> 3, j = m & 7;
        int h = (wh * 8 + i + 4) & 127;
        int w = (ww * 8 + j + 4) & 127;
        size_t idx = ((size_t)(b * 16384 + h * 128 + w)) * C_ + c;
        xres[idx] = x[idx] + acc[r] + pb;
      }
    }
}

// ---------------------------------------------------------------------------
// MLP kernel: 64 rows per block, in-place on xio; 8 waves.
__global__ __launch_bounds__(256)
void k_mlp(const float* __restrict__ n2g, const float* __restrict__ n2b,
           const _Float16* __restrict__ fc1w, const float* __restrict__ fc1b,
           const _Float16* __restrict__ fc2w, const float* __restrict__ fc2b,
           float* __restrict__ xio) {
  extern __shared__ _Float16 lds[];
  _Float16* xln = lds;           // 64*200
  _Float16* hb  = lds + 12800;   // 64*784
  const int wv = threadIdx.x >> 5, lane = threadIdx.x & 31;
  const size_t row0 = (size_t)blockIdx.x * 64;

  // ---- LN2 -> xln f16 ----
  for (int t = wv; t < 64; t += 8) {
    const float* row = xio + (row0 + t) * C_;
    float v0[6];
    float s = 0.f;
#pragma unroll
    for (int kk = 0; kk < 6; ++kk) { v0[kk] = row[kk * 32 + lane]; s += v0[kk]; }
#pragma unroll
    for (int off = 16; off; off >>= 1) s += __shfl_xor(s, off, 32);
    float mean = s * (1.f / 192.f);
    float q2 = 0.f;
#pragma unroll
    for (int kk = 0; kk < 6; ++kk) { float d = v0[kk] - mean; q2 += d * d; }
#pragma unroll
    for (int off = 16; off; off >>= 1) q2 += __shfl_xor(q2, off, 32);
    float rstd = rsqrtf(q2 * (1.f / 192.f) + 1e-5f);
#pragma unroll
    for (int kk = 0; kk < 6; ++kk) {
      int c = kk * 32 + lane;
      xln[t * XSTR + c] = (_Float16)((v0[kk] - mean) * rstd * n2g[c] + n2b[c]);
    }
  }
  __syncthreads();

  // ---- h = gelu(xln @ fc1^T + b), [64,768]; 192 tiles over 8 waves ----
  for (int tid = wv * 24; tid < wv * 24 + 24; ++tid) {
    int nt = tid >> 2, mt = tid & 3;
    v8f acc = {};
#pragma unroll
    for (int ks = 0; ks < 6; ++ks) {
      v16h a = frag_a(xln, XSTR, mt * 16, ks * 32);
      v16h bf = frag_b(fc1w, C_, nt * 16, ks * 32);
      acc = wmma_f16(a, bf, acc);
    }
    int col = nt * 16 + (lane & 15);
    float bb = fc1b[col];
    int m0 = mt * 16 + ((lane >> 4) << 3);
#pragma unroll
    for (int r = 0; r < 8; ++r) {
      float v = acc[r] + bb;
      v = 0.5f * v * (1.f + erff(v * 0.70710678118654752f));  // exact GELU
      hb[(m0 + r) * HSTR + col] = (_Float16)v;
    }
  }
  __syncthreads();

  // ---- out = x + h @ fc2^T + b, in place; 48 tiles over 8 waves ----
  for (int tid = wv * 6; tid < wv * 6 + 6; ++tid) {
    int nt = tid >> 2, mt = tid & 3;
    v8f acc = {};
#pragma unroll
    for (int ks = 0; ks < 24; ++ks) {
      v16h a = frag_a(hb, HSTR, mt * 16, ks * 32);
      v16h bf = frag_b(fc2w, 768, nt * 16, ks * 32);
      acc = wmma_f16(a, bf, acc);
    }
    int c = nt * 16 + (lane & 15);
    float bb = fc2b[c];
    int m0 = mt * 16 + ((lane >> 4) << 3);
#pragma unroll
    for (int r = 0; r < 8; ++r) {
      size_t idx = (row0 + m0 + r) * C_ + c;
      xio[idx] = xio[idx] + acc[r] + bb;   // read original x, then overwrite
    }
  }
}

// ---------------------------------------------------------------------------
extern "C" void kernel_launch(void* const* d_in, const int* in_sizes, int n_in,
                              void* d_out, int out_size, void* d_ws, size_t ws_size,
                              hipStream_t stream) {
  const float* x       = (const float*)d_in[0];
  const int*   rpi     = (const int*)d_in[1];
  const float* amask   = (const float*)d_in[2];
  const float* n1g     = (const float*)d_in[3];
  const float* n1b     = (const float*)d_in[4];
  const float* qkvw_f  = (const float*)d_in[5];
  const float* qkvb    = (const float*)d_in[6];
  const float* gatew_f = (const float*)d_in[7];
  const float* gateb   = (const float*)d_in[8];
  const float* rpb     = (const float*)d_in[9];
  const float* projw_f = (const float*)d_in[10];
  const float* projb   = (const float*)d_in[11];
  const float* n2g     = (const float*)d_in[12];
  const float* n2b     = (const float*)d_in[13];
  const float* fc1w_f  = (const float*)d_in[14];
  const float* fc1b    = (const float*)d_in[15];
  const float* fc2w_f  = (const float*)d_in[16];
  const float* fc2b    = (const float*)d_in[17];

  char* ws = (char*)d_ws;
  _Float16* qkvw  = (_Float16*)(ws + 0);        // 576*192*2 = 221184
  _Float16* gatew = (_Float16*)(ws + 221184);   // 73728
  _Float16* projw = (_Float16*)(ws + 294912);   // 73728
  _Float16* fc1w  = (_Float16*)(ws + 368640);   // 294912
  _Float16* fc2w  = (_Float16*)(ws + 663552);   // 294912
  float*    biasc = (float*)(ws + 958464);      // 256*6*64*64*4 = 25165824

  k_cvt<<<(576 * 192 + 255) / 256, 256, 0, stream>>>(qkvw_f, qkvw, 576 * 192);
  k_cvt<<<(192 * 192 + 255) / 256, 256, 0, stream>>>(gatew_f, gatew, 192 * 192);
  k_cvt<<<(192 * 192 + 255) / 256, 256, 0, stream>>>(projw_f, projw, 192 * 192);
  k_cvt<<<(768 * 192 + 255) / 256, 256, 0, stream>>>(fc1w_f, fc1w, 768 * 192);
  k_cvt<<<(192 * 768 + 255) / 256, 256, 0, stream>>>(fc2w_f, fc2w, 192 * 768);
  k_bias<<<(256 * 4096 + 255) / 256, 256, 0, stream>>>(amask, rpi, rpb, biasc);

  // attention: 2048 windows; LDS = (39424 + 6*64*72) halves = 134144 bytes
  k_attn<<<2048, 192, 134144, stream>>>(x, n1g, n1b, qkvw, qkvb, gatew, gateb,
                                        projw, projb, biasc, (float*)d_out);
  // MLP: 131072 rows / 64 = 2048 blocks; LDS = (12800 + 64*784)*2 = 125952 bytes
  k_mlp<<<2048, 256, 125952, stream>>>(n2g, n2b, fc1w, fc1b, fc2w, fc2b,
                                       (float*)d_out);
}
